// Hausdorff_77515569758538
// MI455X (gfx1250) — compile-verified, matrix operand order checked by reading, not run on hardware
//
#include <hip/hip_runtime.h>
#include <hip/hip_bf16.h>

typedef __attribute__((ext_vector_type(8))) int v8i;

#define IMG_PIX 65536   // 256*256
#define NB 8
#define NC 3            // classes 1..3 (background excluded)
#define NIMG (NB * NC)

// -------------------------------------------------------------------------
// Kernel 1: channel argmax (first-max tie-break, matching jnp.argmax)
// pred/ref: (8, 4, 256, 256) fp32 -> u8 labels (8, 256, 256)
// -------------------------------------------------------------------------
__global__ void hd_argmax_kernel(const float* __restrict__ pred,
                                 const float* __restrict__ ref,
                                 unsigned char* __restrict__ pl,
                                 unsigned char* __restrict__ rl) {
    int g = blockIdx.x * blockDim.x + threadIdx.x;
    if (g >= NB * IMG_PIX) return;
    int b = g >> 16;
    int p = g & (IMG_PIX - 1);
    const float* pb = pred + (size_t)b * 4 * IMG_PIX + p;
    const float* rb = ref  + (size_t)b * 4 * IMG_PIX + p;

    float bv = pb[0]; int bi = 0;
#pragma unroll
    for (int ch = 1; ch < 4; ++ch) { float v = pb[ch * IMG_PIX]; if (v > bv) { bv = v; bi = ch; } }
    pl[g] = (unsigned char)bi;

    bv = rb[0]; bi = 0;
#pragma unroll
    for (int ch = 1; ch < 4; ++ch) { float v = rb[ch * IMG_PIX]; if (v > bv) { bv = v; bi = ch; } }
    rl[g] = (unsigned char)bi;
}

// -------------------------------------------------------------------------
// Kernel 2: hasref via WMMA IU8 popcount.
// One wave (32 lanes) per (b,c). Each lane feeds 32 mask bytes/chunk into the
// A matrix; B = all-ones. C[m][n] = rowsum_m, so sum(all C) = 16 * popcount.
// Two independent accumulators break the IU8 RAW (D->C) hazard chain so the
// byte-compare VALU work co-executes under each v_wmma instead of v_nops.
// -------------------------------------------------------------------------
__device__ __forceinline__ unsigned eq_bytes01(unsigned v, unsigned cc) {
    // per-byte: (byte == c) ? 0x01 : 0x00
    unsigned x = v ^ cc;
    unsigned t = (((x & 0x7F7F7F7Fu) + 0x7F7F7F7Fu) | x) | 0x7F7F7F7Fu;
    return ((~t) >> 7) & 0x01010101u;
}

__device__ __forceinline__ v8i mask_chunk(const unsigned char* p, unsigned cc) {
    const uint4* q = (const uint4*)p;
    uint4 w0 = q[0];
    uint4 w1 = q[1];
    v8i a;
    a[0] = (int)eq_bytes01(w0.x, cc); a[1] = (int)eq_bytes01(w0.y, cc);
    a[2] = (int)eq_bytes01(w0.z, cc); a[3] = (int)eq_bytes01(w0.w, cc);
    a[4] = (int)eq_bytes01(w1.x, cc); a[5] = (int)eq_bytes01(w1.y, cc);
    a[6] = (int)eq_bytes01(w1.z, cc); a[7] = (int)eq_bytes01(w1.w, cc);
    return a;
}

__global__ void hd_hasref_wmma_kernel(const unsigned char* __restrict__ ref_lbl,
                                      float* __restrict__ hasref) {
    int img = blockIdx.x;                 // 0..23
    int b = img / NC;
    unsigned c = (unsigned)(img % NC) + 1u;
    int lane = threadIdx.x;               // 0..31 (wave32)
    const unsigned char* base = ref_lbl + (size_t)b * IMG_PIX;
    unsigned cc = 0x01010101u * c;

    v8i ones = {0x01010101, 0x01010101, 0x01010101, 0x01010101,
                0x01010101, 0x01010101, 0x01010101, 0x01010101};
    v8i acc0 = {0, 0, 0, 0, 0, 0, 0, 0};
    v8i acc1 = {0, 0, 0, 0, 0, 0, 0, 0};

    // 64 chunks of 1024B; two in flight per iteration on separate accumulators
    for (int i = 0; i < 64; i += 2) {
        v8i a0 = mask_chunk(base + (i + 0) * 1024 + lane * 32, cc);
        v8i a1 = mask_chunk(base + (i + 1) * 1024 + lane * 32, cc);
        acc0 = __builtin_amdgcn_wmma_i32_16x16x64_iu8(false, a0, false, ones,
                                                      acc0, false, false);
        acc1 = __builtin_amdgcn_wmma_i32_16x16x64_iu8(false, a1, false, ones,
                                                      acc1, false, false);
    }
    int part = 0;
#pragma unroll
    for (int k = 0; k < 8; ++k) part += acc0[k] + acc1[k];
#pragma unroll
    for (int off = 16; off > 0; off >>= 1) part += __shfl_xor(part, off, 32);
    if (lane == 0) hasref[img] = (part > 0) ? 1.0f : 0.0f;   // part == 16*count
}

// -------------------------------------------------------------------------
// Kernel 3: per-(b,c) exact EDT + directed Hausdorff, fully in 256KB LDS.
// Phase A: column pass (thread = column): border-on-the-fly, two 1D sweeps,
//          store dy^2 (BIG ~ 1e12 where column has no border).
// Phase B: row pass (thread = output x): brute-force min-plus over the LDS
//          row via float4 broadcasts (ds_load_b128), sqrt, mask by pred
//          border, running max; block max-reduce.
// -------------------------------------------------------------------------
__device__ __forceinline__ bool border_at(const unsigned char* __restrict__ img,
                                          unsigned c, int y, int x) {
    int idx = y * 256 + x;
    if (img[idx] != c) return false;
    if (y == 0 || y == 255 || x == 0 || x == 255) return true;
    return (img[idx - 256] != c) | (img[idx + 256] != c) |
           (img[idx - 1] != c)   | (img[idx + 1] != c);
}

__global__ void hd_edt_kernel(const unsigned char* __restrict__ pred_lbl,
                              const unsigned char* __restrict__ ref_lbl,
                              float* __restrict__ h) {
    extern __shared__ float sm[];         // 65536 floats = 256 KB (CDNA5 LDS)
    int img = blockIdx.x;
    int b = img / NC;
    unsigned c = (unsigned)(img % NC) + 1u;
    int x = threadIdx.x;                  // 0..255, thread owns column x

    const unsigned char* refimg  = ref_lbl  + (size_t)b * IMG_PIX;
    const unsigned char* predimg = pred_lbl + (size_t)b * IMG_PIX;

    // ---- Phase A: 1D EDT along H (axis 1) per column ----
    unsigned bb[8] = {0, 0, 0, 0, 0, 0, 0, 0};  // border bits cache
    float run = 1.0e6f;                          // run^2 == 1e12 == BIG
    for (int y = 0; y < 256; ++y) {
        bool br = border_at(refimg, c, y, x);
        if (br) { run = 0.0f; bb[y >> 5] |= (1u << (y & 31)); }
        else    { run += 1.0f; }
        sm[y * 256 + x] = run * run;
    }
    run = 1.0e6f;
    for (int y = 255; y >= 0; --y) {
        bool br = (bb[y >> 5] >> (y & 31)) & 1u;
        if (br) run = 0.0f; else run += 1.0f;
        int idx = y * 256 + x;
        float v = run * run;
        sm[idx] = fminf(sm[idx], v);
    }
    __syncthreads();

    // ---- Phase B: min-plus along W (axis 2), fused sqrt+mask+max ----
    float mymax = 0.0f;                   // matches max(where(bp, dist, 0))
    for (int y = 0; y < 256; ++y) {
        __builtin_prefetch(predimg + ((y + 1) & 255) * 256 + x, 0, 1);
        const float4* row4 = (const float4*)(sm + y * 256);  // LDS b128 bcast
        float best = 3.0e38f;
#pragma unroll 4
        for (int q = 0; q < 64; ++q) {
            float4 dv = row4[q];
            int xp = q * 4;
            int d0 = x - xp, d1 = x - xp - 1, d2 = x - xp - 2, d3 = x - xp - 3;
            float c0 = dv.x + (float)(d0 * d0);
            float c1 = dv.y + (float)(d1 * d1);
            float c2 = dv.z + (float)(d2 * d2);
            float c3 = dv.w + (float)(d3 * d3);
            best = fminf(best, fminf(fminf(c0, c1), fminf(c2, c3)));
        }
        bool bp = border_at(predimg, c, y, x);
        float val = bp ? sqrtf(best) : 0.0f;
        mymax = fmaxf(mymax, val);
    }

    // ---- block max-reduce (reuse LDS) ----
    __syncthreads();
    sm[x] = mymax;
    __syncthreads();
    for (int s = 128; s > 0; s >>= 1) {
        if (x < s) sm[x] = fmaxf(sm[x], sm[x + s]);
        __syncthreads();
    }
    if (x == 0) h[img] = sm[0];
}

// -------------------------------------------------------------------------
// Kernel 4: finalize -> scalar mean over batch of (sum_c gated_h / sum_c w)
// -------------------------------------------------------------------------
__global__ void hd_finalize_kernel(const float* __restrict__ h,
                                   const float* __restrict__ hasref,
                                   float* __restrict__ out) {
    if (blockIdx.x == 0 && threadIdx.x == 0) {
        float acc = 0.0f;
        for (int b = 0; b < NB; ++b) {
            float loss = 0.0f, sw = 0.0f;
            for (int ci = 0; ci < NC; ++ci) {
                int img = b * NC + ci;
                float w = hasref[img];
                loss += w * h[img];
                sw += w;
            }
            acc += loss / sw;             // same div-by-zero behavior as ref
        }
        out[0] = acc / (float)NB;
    }
}

// -------------------------------------------------------------------------
extern "C" void kernel_launch(void* const* d_in, const int* in_sizes, int n_in,
                              void* d_out, int out_size, void* d_ws, size_t ws_size,
                              hipStream_t stream) {
    (void)in_sizes; (void)n_in; (void)out_size; (void)ws_size;
    const float* pred = (const float*)d_in[0];
    const float* ref  = (const float*)d_in[1];
    float* out = (float*)d_out;

    // workspace layout (~1 MB)
    unsigned char* pred_lbl = (unsigned char*)d_ws;                 // 524288 B
    unsigned char* ref_lbl  = pred_lbl + (size_t)NB * IMG_PIX;      // 524288 B
    float* h      = (float*)(ref_lbl + (size_t)NB * IMG_PIX);       // 24 floats
    float* hasref = h + 32;                                         // 24 floats

    int total = NB * IMG_PIX;
    hd_argmax_kernel<<<(total + 255) / 256, 256, 0, stream>>>(pred, ref,
                                                              pred_lbl, ref_lbl);
    hd_hasref_wmma_kernel<<<NIMG, 32, 0, stream>>>(ref_lbl, hasref);
    hd_edt_kernel<<<NIMG, 256, IMG_PIX * sizeof(float), stream>>>(pred_lbl,
                                                                  ref_lbl, h);
    hd_finalize_kernel<<<1, 32, 0, stream>>>(h, hasref, out);
}
